// AdvancedComRoPEAttention_7739531067593
// MI455X (gfx1250) — matrix operand order repaired and checked by reading, not compiled, not run on hardware
//
#include <hip/hip_runtime.h>
#include <hip/hip_bf16.h>

// ---------------------------------------------------------------------------
// Types for CDNA5 WMMA (wave32): v_wmma_f32_16x16x32_bf16
// ---------------------------------------------------------------------------
typedef __bf16 v16bf __attribute__((ext_vector_type(16)));
typedef float  v8f   __attribute__((ext_vector_type(8)));
typedef unsigned int v4u __attribute__((ext_vector_type(4)));
typedef int          v8i __attribute__((ext_vector_type(8)));
typedef int          v4i __attribute__((ext_vector_type(4)));

union FragBF { v16bf v; unsigned short u[16]; };

static __device__ __forceinline__ unsigned short f32_to_bf16(float f) {
    unsigned int u = __float_as_uint(f);
    unsigned int r = u + 0x7FFFu + ((u >> 16) & 1u);   // round-to-nearest-even
    return (unsigned short)(r >> 16);
}
static __device__ __forceinline__ float bf16_to_f32(unsigned short h) {
    return __uint_as_float(((unsigned int)h) << 16);
}

// ---------------------------------------------------------------------------
// Tensor Data Mover staging (gfx1250). 2-D tile load: [rows x cols] bf16,
// row stride `ld` elements, packed row-major into LDS at lds_byte_off.
// D# packing per CDNA5 ISA (group0/group1 bitfields).
// This toolchain exposes the 6-arg builtin:
//   (uint32x4 g0, int32x8 g1, int32x4 g2, int32x4 g3, int32x8 g4, i32 cpol)
// ---------------------------------------------------------------------------
#if defined(__gfx1250__) && __has_builtin(__builtin_amdgcn_tensor_load_to_lds) && \
    __has_builtin(__builtin_amdgcn_s_wait_tensorcnt)
#define USE_TDM 1
#else
#define USE_TDM 0
#endif

typedef __attribute__((address_space(3))) unsigned short lds_ushort;

#if USE_TDM
static __device__ __forceinline__ void tdm_load_tile_2d_bf16(
        const unsigned short* gsrc, int row0, int k0, int ld,
        unsigned int lds_byte_off, unsigned int rows, unsigned int cols) {
    unsigned long long ga = (unsigned long long)(const unsigned char*)gsrc +
        ((unsigned long long)(unsigned)row0 * (unsigned long long)(unsigned)ld +
         (unsigned long long)(unsigned)k0) * 2ull;
    v4u g0;
    g0.x = 1u;                                              // count=1, user mode
    g0.y = lds_byte_off;                                    // lds_addr
    g0.z = (unsigned int)(ga & 0xFFFFFFFFull);              // global_addr[31:0]
    g0.w = (unsigned int)((ga >> 32) & 0x1FFFFFFull) | (2u << 30); // addr hi | type=2
    unsigned long long stride = (unsigned long long)(unsigned)ld;  // elements
    v8i g1;
    g1[0] = (int)(1u << 16);                                // data_size=1 (2 bytes)
    g1[1] = (int)((cols & 0xFFFFu) << 16);                  // tensor_dim0[15:0]
    g1[2] = (int)(((cols >> 16) & 0xFFFFu) | ((rows & 0xFFFFu) << 16)); // d0 hi|d1 lo
    g1[3] = (int)(((rows >> 16) & 0xFFFFu) | ((cols & 0xFFFFu) << 16)); // d1 hi|tile_dim0
    g1[4] = (int)(rows & 0xFFFFu);                          // tile_dim1 (tile_dim2=0)
    g1[5] = (int)(unsigned int)(stride & 0xFFFFFFFFull);    // dim0_stride[31:0]
    g1[6] = (int)(unsigned int)((stride >> 32) & 0xFFFFull);// dim0_stride[47:32]
    g1[7] = 0;
    v4i z4 = {0, 0, 0, 0};
    v8i z8 = {0, 0, 0, 0, 0, 0, 0, 0};
    __builtin_amdgcn_tensor_load_to_lds(g0, g1, z4, z4, z8, 0);
}
#endif

// ---------------------------------------------------------------------------
// fp32 -> bf16 elementwise conversion
// ---------------------------------------------------------------------------
__global__ void cvt_f32_to_bf16_kernel(const float* __restrict__ src,
                                       unsigned short* __restrict__ dst, int n) {
    int i = blockIdx.x * blockDim.x + threadIdx.x;
    if (i < n) dst[i] = f32_to_bf16(src[i]);
}

// ---------------------------------------------------------------------------
// bmean[s] = mean(bias[s, :]) * 0.1
// ---------------------------------------------------------------------------
__global__ void bias_row_mean_kernel(const float* __restrict__ bias,
                                     float* __restrict__ bm, int S) {
    __shared__ float red[256];
    int s = blockIdx.x;
    float acc = 0.f;
    for (int j = threadIdx.x; j < S; j += blockDim.x)
        acc += bias[(size_t)s * S + j];
    red[threadIdx.x] = acc;
    __syncthreads();
    for (int off = 128; off > 0; off >>= 1) {
        if ((int)threadIdx.x < off) red[threadIdx.x] += red[threadIdx.x + off];
        __syncthreads();
    }
    if (threadIdx.x == 0) bm[s] = red[0] / (float)S * 0.1f;
}

// ---------------------------------------------------------------------------
// RoPE + bmean, in place on bf16 Q/K stored as [B, S, H*D]
// ---------------------------------------------------------------------------
__global__ void rope_bias_kernel(unsigned short* __restrict__ qh,
                                 unsigned short* __restrict__ kh,
                                 const float* __restrict__ bm,
                                 int B, int S, int H, int D) {
    int idx = blockIdx.x * blockDim.x + threadIdx.x;
    int half = D >> 1;
    int total = B * S * H * half;
    if (idx >= total) return;
    int i = idx % half;
    int h = (idx / half) % H;
    int s = (idx / (half * H)) % S;
    int b = idx / (half * H * S);

    float invf = __powf(10000.0f, -(float)i / (float)half);   // base^(-(2i)/D)
    float ang  = (float)s * invf;
    float c = __cosf(ang), sn = __sinf(ang);
    float bmv = bm[s];

    size_t base = ((size_t)(b * S + s)) * (size_t)(H * D) + (size_t)h * D;

    float q0 = bf16_to_f32(qh[base + i]);
    float q1 = bf16_to_f32(qh[base + i + half]);
    qh[base + i]        = f32_to_bf16(q0 * c - q1 * sn + bmv);
    qh[base + i + half] = f32_to_bf16(q1 * c + q0 * sn + bmv);

    float k0 = bf16_to_f32(kh[base + i]);
    float k1 = bf16_to_f32(kh[base + i + half]);
    kh[base + i]        = f32_to_bf16(k0 * c - k1 * sn + bmv);
    kh[base + i + half] = f32_to_bf16(k1 * c + k0 * sn + bmv);
}

// ---------------------------------------------------------------------------
// Blocked NT GEMM: C[m,n] = scale * sum_k A[m,k] * B[n,k]
//   A_F32 : A is fp32, staged cooperatively into LDS with bf16 convert
//           (else bf16, staged via Tensor Data Mover)
//   C_F32 : C written fp32, else bf16
//   cTrans: store C transposed (C[n,m], ldc = rows of the transposed view)
//   causalSkip: drop block tiles entirely above the diagonal (QK^T scores)
//   causalKlim: stop the K loop at mBase+64 (P rows are causal-zero beyond)
// Block = 128 threads = 4 waves in a 2x2 arrangement; block tile 64x64,
// per-wave tile 32x32 (2x2 accumulators -> 4 WMMAs per K-step).
// ---------------------------------------------------------------------------
template <bool A_F32, bool C_F32>
__global__ void wmma_gemm_nt_kernel(const void* __restrict__ Aall,
                                    const unsigned short* __restrict__ Ball,
                                    void* __restrict__ Call,
                                    int M, int N, int K,
                                    int lda, int ldb, int ldc,
                                    size_t aSB, size_t aSH,
                                    size_t bSB, size_t bSH,
                                    size_t cSB, size_t cSH,
                                    int HB, float scale,
                                    int cTrans, int causalSkip, int causalKlim) {
    __shared__ unsigned short ldsA[64 * 32];
    __shared__ unsigned short ldsB[64 * 32];

    int mBase = blockIdx.x * 64;
    int nBase = blockIdx.y * 64;
    if (causalSkip && nBase > mBase + 63) return;   // fully masked score tile
    if (mBase >= M || nBase >= N) return;

    int zb = blockIdx.z / HB;
    int zh = blockIdx.z % HB;
    size_t aOff = aSB * zb + aSH * zh;
    const unsigned short* Ab = (const unsigned short*)Aall + aOff;
    const float*          Af = (const float*)Aall + aOff;
    const unsigned short* Bp = Ball + bSB * zb + bSH * zh;
    size_t cOff = cSB * zb + cSH * zh;

    int tid  = threadIdx.x;
    int wave = tid >> 5;
    int lane = tid & 31;
    int wm = wave >> 1;          // 0..1 : which 32-row half of the block tile
    int wn = wave & 1;           // 0..1 : which 32-col half

    int kbaseA = (lane >> 4) * 8;    // A 16x32 layout: hi lanes hold K+8 / K+24
    int kgrpB  = (lane >> 4) * 8;    // B 32x16 layout: hi lanes hold K+8 / K+24

#if USE_TDM
    unsigned int ldsA_off = (unsigned int)(unsigned long long)(lds_ushort*)ldsA;
    unsigned int ldsB_off = (unsigned int)(unsigned long long)(lds_ushort*)ldsB;
#endif

    v8f acc[2][2];
#pragma unroll
    for (int i = 0; i < 2; ++i)
#pragma unroll
        for (int j = 0; j < 2; ++j)
            acc[i][j] = (v8f){0.f, 0.f, 0.f, 0.f, 0.f, 0.f, 0.f, 0.f};

    int Kend = causalKlim ? (mBase + 64 < K ? mBase + 64 : K) : K;

    int srow = tid >> 1;            // 0..63  (cooperative staging position)
    int sseg = (tid & 1) * 16;      // two 16-element segments per row

    for (int k0 = 0; k0 < Kend; k0 += 32) {
        // ---- stage 64x32 A and B tiles into LDS ----
#if USE_TDM
        if (A_F32) {
            // fp32 A: coalesced loads + convert; B via Tensor Data Mover
#pragma unroll
            for (int j = 0; j < 16; ++j)
                ldsA[srow * 32 + sseg + j] =
                    f32_to_bf16(Af[(size_t)(mBase + srow) * lda + k0 + sseg + j]);
            if (tid < 32) {
                tdm_load_tile_2d_bf16(Bp, nBase, k0, ldb, ldsB_off, 64u, 32u);
                __builtin_amdgcn_s_wait_tensorcnt(0);
            }
        } else {
            if (tid < 32) {   // wave 0 drives the Tensor Data Mover
                tdm_load_tile_2d_bf16(Ab, mBase, k0, lda, ldsA_off, 64u, 32u);
                tdm_load_tile_2d_bf16(Bp, nBase, k0, ldb, ldsB_off, 64u, 32u);
                __builtin_amdgcn_s_wait_tensorcnt(0);
            }
        }
#else
        if (A_F32) {
#pragma unroll
            for (int j = 0; j < 16; ++j)
                ldsA[srow * 32 + sseg + j] =
                    f32_to_bf16(Af[(size_t)(mBase + srow) * lda + k0 + sseg + j]);
        } else {
#pragma unroll
            for (int j = 0; j < 16; ++j)
                ldsA[srow * 32 + sseg + j] =
                    Ab[(size_t)(mBase + srow) * lda + k0 + sseg + j];
        }
#pragma unroll
        for (int j = 0; j < 16; ++j)
            ldsB[srow * 32 + sseg + j] =
                Bp[(size_t)(nBase + srow) * ldb + k0 + sseg + j];
#endif
        __syncthreads();

        // ---- build fragments from LDS, 4 WMMAs per K-step ----
        FragBF a[2], b[2];
#pragma unroll
        for (int sm = 0; sm < 2; ++sm) {
            int mloc = wm * 32 + sm * 16 + (lane & 15);
#pragma unroll
            for (int e = 0; e < 16; ++e)
                a[sm].u[e] = ldsA[mloc * 32 + kbaseA + (e & 7) + ((e >> 3) << 4)];
        }
#pragma unroll
        for (int sn = 0; sn < 2; ++sn) {
            int nloc = wn * 32 + sn * 16 + (lane & 15);
#pragma unroll
            for (int e = 0; e < 16; ++e)
                b[sn].u[e] = ldsB[nloc * 32 + kgrpB + (e & 7) + ((e >> 3) << 4)];
        }
#pragma unroll
        for (int sm = 0; sm < 2; ++sm)
#pragma unroll
            for (int sn = 0; sn < 2; ++sn)
                acc[sm][sn] = __builtin_amdgcn_wmma_f32_16x16x32_bf16(
                    false, a[sm].v, false, b[sn].v, (short)0, acc[sm][sn],
                    false, false);
        __syncthreads();
    }

    // ---- epilogue: 32x32 per-wave tile (optionally transposed store) ----
#pragma unroll
    for (int sm = 0; sm < 2; ++sm)
#pragma unroll
        for (int sn = 0; sn < 2; ++sn)
#pragma unroll
            for (int r = 0; r < 8; ++r) {
                int m = mBase + wm * 32 + sm * 16 + r + ((lane >> 4) << 3);
                int n = nBase + wn * 32 + sn * 16 + (lane & 15);
                float val = acc[sm][sn][r] * scale;
                size_t ci = cTrans ? (cOff + (size_t)n * ldc + m)
                                   : (cOff + (size_t)m * ldc + n);
                if (C_F32) ((float*)Call)[ci] = val;
                else       ((unsigned short*)Call)[ci] = f32_to_bf16(val);
            }
}

// ---------------------------------------------------------------------------
// Row softmax with additive bias and causal mask, in place on attn_weights.
// ---------------------------------------------------------------------------
__global__ void softmax_bias_kernel(float* __restrict__ attnw,
                                    const float* __restrict__ bias, int S) {
    __shared__ float red[256];
    size_t row = blockIdx.x;
    int q = (int)(row % (size_t)S);
    float* p = attnw + row * (size_t)S;
    const float* brow = bias + (size_t)q * S;

    float mx = -3.4e38f;
    for (int k = threadIdx.x; k <= q; k += blockDim.x)
        mx = fmaxf(mx, p[k] + brow[k]);
    red[threadIdx.x] = mx;
    __syncthreads();
    for (int off = 128; off > 0; off >>= 1) {
        if ((int)threadIdx.x < off)
            red[threadIdx.x] = fmaxf(red[threadIdx.x], red[threadIdx.x + off]);
        __syncthreads();
    }
    mx = red[0];
    __syncthreads();

    float sum = 0.f;
    for (int k = threadIdx.x; k <= q; k += blockDim.x)
        sum += __expf(p[k] + brow[k] - mx);
    red[threadIdx.x] = sum;
    __syncthreads();
    for (int off = 128; off > 0; off >>= 1) {
        if ((int)threadIdx.x < off) red[threadIdx.x] += red[threadIdx.x + off];
        __syncthreads();
    }
    float inv = 1.0f / red[0];

    for (int k = threadIdx.x; k < S; k += blockDim.x) {
        float v = (k <= q) ? __expf(p[k] + brow[k] - mx) * inv : 0.0f;
        p[k] = v;
    }
}

// ---------------------------------------------------------------------------
// Host-side orchestration
// ---------------------------------------------------------------------------
extern "C" void kernel_launch(void* const* d_in, const int* in_sizes, int n_in,
                              void* d_out, int out_size, void* d_ws, size_t ws_size,
                              hipStream_t stream) {
    const int B = 2, S = 2048, HID = 2048, H = 16, D = 128;

    const float* x    = (const float*)d_in[0];
    const float* Wq   = (const float*)d_in[1];
    const float* Wk   = (const float*)d_in[2];
    const float* Wv   = (const float*)d_in[3];
    const float* Wo   = (const float*)d_in[4];
    const float* bias = (const float*)d_in[5];

    float* out_proj = (float*)d_out;                         // [B,S,HID]
    float* attnw    = out_proj + (size_t)B * S * HID;        // [B,H,S,S]

    char*  ws  = (char*)d_ws;
    size_t off = 0;
    auto carve = [&](size_t bytes) -> void* {
        void* p = ws + off;
        off += (bytes + 255) & ~(size_t)255;
        return p;
    };
    const size_t nX = (size_t)B * S * HID;   // 8,388,608
    const size_t nW = (size_t)HID * HID;     // 4,194,304
    unsigned short* xb  = (unsigned short*)carve(nX * 2);
    unsigned short* wqb = (unsigned short*)carve(nW * 2);
    unsigned short* wkb = (unsigned short*)carve(nW * 2);
    unsigned short* wvb = (unsigned short*)carve(nW * 2);
    unsigned short* wob = (unsigned short*)carve(nW * 2);
    unsigned short* qh  = (unsigned short*)carve(nX * 2);    // [B,S,H*D]
    unsigned short* kh  = (unsigned short*)carve(nX * 2);
    unsigned short* vt  = (unsigned short*)carve(nX * 2);    // V^T  [B,H,D,S]
    unsigned short* aoh = (unsigned short*)carve(nX * 2);    // attn_out [B,S,H*D]
    float* bm = (float*)carve(S * sizeof(float));

    // 1) fp32 -> bf16 conversions
    cvt_f32_to_bf16_kernel<<<((int)nX + 255) / 256, 256, 0, stream>>>(x, xb, (int)nX);
    cvt_f32_to_bf16_kernel<<<((int)nW + 255) / 256, 256, 0, stream>>>(Wq, wqb, (int)nW);
    cvt_f32_to_bf16_kernel<<<((int)nW + 255) / 256, 256, 0, stream>>>(Wk, wkb, (int)nW);
    cvt_f32_to_bf16_kernel<<<((int)nW + 255) / 256, 256, 0, stream>>>(Wv, wvb, (int)nW);
    cvt_f32_to_bf16_kernel<<<((int)nW + 255) / 256, 256, 0, stream>>>(Wo, wob, (int)nW);

    // 2) bias row means
    bias_row_mean_kernel<<<S, 256, 0, stream>>>(bias, bm, S);

    // 3) Q/K projections: y = x @ W^T  (NT, bf16 out, [B,S,H*D] layout)
    dim3 gp(B * S / 64, HID / 64, 1);   // (64, 32, 1)
    wmma_gemm_nt_kernel<false, false><<<gp, 128, 0, stream>>>(
        xb, wqb, qh, B * S, HID, HID, HID, HID, HID,
        0, 0, 0, 0, 0, 0, 1, 1.0f, 0, 0, 0);
    wmma_gemm_nt_kernel<false, false><<<gp, 128, 0, stream>>>(
        xb, wkb, kh, B * S, HID, HID, HID, HID, HID,
        0, 0, 0, 0, 0, 0, 1, 1.0f, 0, 0, 0);
    // V projection with transposed store -> vt[b, h, d, s]  (batched over b)
    dim3 gv(S / 64, HID / 64, B);       // (32, 32, 2)
    wmma_gemm_nt_kernel<false, false><<<gv, 128, 0, stream>>>(
        xb, wvb, vt, S, HID, HID, HID, HID, S,
        (size_t)S * HID, 0,                    // A: per-batch row block of x
        0, 0,                                  // B: shared weights
        (size_t)HID * S, 0,                    // C: per-batch V^T block
        1, 1.0f, /*cTrans=*/1, 0, 0);

    // 4) RoPE + bmean on Q,K
    int nrope = B * S * H * (D / 2);
    rope_bias_kernel<<<(nrope + 255) / 256, 256, 0, stream>>>(qh, kh, bm, B, S, H, D);

    // 5) scores = (Q @ K^T) / sqrt(D), batched over (b,h), causal tiles skipped
    dim3 gs(S / 64, S / 64, B * H);     // (32, 32, 32)
    wmma_gemm_nt_kernel<false, true><<<gs, 128, 0, stream>>>(
        qh, kh, attnw, S, S, D, HID, HID, S,
        (size_t)S * HID, (size_t)D,            // A: b-stride, h-stride
        (size_t)S * HID, (size_t)D,            // B
        (size_t)H * S * S, (size_t)S * S,      // C
        H, 0.08838834764831845f, 0, /*causalSkip=*/1, 0);

    // 6) bias + causal mask + softmax, in place on attn_weights
    softmax_bias_kernel<<<B * H * S, 256, 0, stream>>>(attnw, bias, S);

    // 7) attn_out = P @ V = P @ (V^T)^T : NT GEMM, fp32 A, causal K limit
    dim3 ga(S / 64, D / 64, B * H);     // (32, 2, 32)
    wmma_gemm_nt_kernel<true, false><<<ga, 128, 0, stream>>>(
        attnw, vt, aoh, S, D, S, S, S, HID,
        (size_t)H * S * S, (size_t)S * S,      // A (fp32 P)
        (size_t)HID * S, (size_t)D * S,        // B (V^T per b,h)
        (size_t)S * HID, (size_t)D,            // C (aoh [B,S,H*D])
        H, 1.0f, 0, 0, /*causalKlim=*/1);

    // 8) output = attn_out @ Wo^T (NT, fp32 out)
    wmma_gemm_nt_kernel<false, true><<<gp, 128, 0, stream>>>(
        aoh, wob, out_proj, B * S, HID, HID, HID, HID, HID,
        0, 0, 0, 0, 0, 0, 1, 1.0f, 0, 0, 0);

    (void)in_sizes; (void)n_in; (void)out_size; (void)ws_size;
}
